// CostVolume_46153718563219
// MI455X (gfx1250) — matrix operand order repaired and checked by reading, not compiled, not run on hardware
//
#include <hip/hip_runtime.h>

// MI455X / gfx1250: wave32, WMMA f32<-f16 16x16x32 path.
// Pipeline: [zero ws] -> [transpose L/R to f16 NHWC] -> [repack weights f16 + fold BN]
//           -> [cost volume as banded WMMA GEMM] -> [4x implicit-GEMM WMMA conv]
// Conv waves compute an MT x NT register grid of 16x16 tiles (MT*NT WMMA per
// MT+NT operand fetches) to raise FLOP/byte from the cache hierarchy
// (2x4 tile: 8 WMMA / 6KB operands = ~21 FLOP/B from L0).

typedef __attribute__((ext_vector_type(16))) _Float16 v16h;
typedef __attribute__((ext_vector_type(8)))  _Float16 v8h;
typedef __attribute__((ext_vector_type(8)))  float    v8f;

#define HH 128
#define WW 256
#define CC 32
#define DD 200
#define HP 130   // H + 2 (zero border ring)
#define WP 258   // W + 2
#define DPAD 224 // 200 padded to 7*32 chunks

// ISA 7.12.2 operand layouts (wave32, 16-bit A 16x32; B 32x16; C/D 16x16 f32)
__device__ __forceinline__ int klocA(int r, int g) {
  // lane = M (lane&15); VGPR r holds K pair; lanes 16-31 offset by 8
  return (r < 4) ? (2 * r + 8 * g) : (2 * r + 8 + 8 * g);
}
__device__ __forceinline__ int klocB(int r, int g) {
  // lane = N (lane&15); VGPR r holds K pair; lanes 16-31 offset by 8
  return (r < 4) ? (2 * r + 8 * g) : (2 * r + 8 + 8 * g);
}

// ---------------------------------------------------------------- zero
__global__ void zero_kernel(float4* __restrict__ p, long n16) {
  long i = (long)blockIdx.x * blockDim.x + threadIdx.x;
  long stride = (long)gridDim.x * blockDim.x;
  float4 z = {0.f, 0.f, 0.f, 0.f};
  for (; i < n16; i += stride) p[i] = z;
}

// ------------------------------------------- L/R f32 NCHW -> f16 [y][x][c]
__global__ void cvt_lr(const float* __restrict__ L, const float* __restrict__ R,
                       _Float16* __restrict__ Lt, _Float16* __restrict__ Rt) {
  int i = blockIdx.x * blockDim.x + threadIdx.x;
  const int total = CC * HH * WW;
  if (i >= total) return;
  int x = i % WW;
  int y = (i / WW) % HH;
  int c = i / (WW * HH);
  int o = (y * WW + x) * CC + c;
  Lt[o] = (_Float16)L[i];
  Rt[o] = (_Float16)R[i];
}

// ------------- repack conv weights [O][Cin][3][3] f32 -> [Opad][9*Cpad] f16,
// ------------- fold conv bias + BN into per-channel scale/shift
__global__ void prep_w(const float* __restrict__ w, const float* __restrict__ b,
                       const float* __restrict__ gg, const float* __restrict__ be,
                       const float* __restrict__ mm, const float* __restrict__ vv,
                       _Float16* __restrict__ W2, float* __restrict__ sc,
                       float* __restrict__ sh, int O, int Opad, int Cin, int Cpad,
                       int has_bn) {
  int total = Opad * 9 * Cpad;
  int i = blockIdx.x * blockDim.x + threadIdx.x;
  int stride = gridDim.x * blockDim.x;
  for (int idx = i; idx < total; idx += stride) {
    int ci = idx % Cpad;
    int kykx = (idx / Cpad) % 9;
    int o = idx / (9 * Cpad);
    float val = 0.f;
    if (o < O && ci < Cin) val = w[(o * Cin + ci) * 9 + kykx];
    W2[idx] = (_Float16)val;
  }
  if (i < Opad) {
    float s = 0.f, t = 0.f;
    if (i < O) {
      if (has_bn) {
        s = gg[i] * rsqrtf(vv[i] + 1e-5f);
        t = s * (b[i] - mm[i]) + be[i];
      } else {
        s = 1.f;
        t = b[i];
      }
    }
    sc[i] = s;
    sh[i] = t;
  }
}

// ---------------- cost volume as banded GEMM: one 16x16x32 WMMA per tile pair
// cost[x, xr] = sum_c L[c,x]*R[c,xr],  d = xr - x + 200, keep 0 <= d < 200
__global__ void corr_wmma(const _Float16* __restrict__ Lt,
                          const _Float16* __restrict__ Rt,
                          _Float16* __restrict__ vol) {
  int y = blockIdx.x;                 // 0..127
  int wave = threadIdx.x >> 5;        // 8 waves / block
  int lane = threadIdx.x & 31;
  int g = lane >> 4, mn = lane & 15;
  int xt = blockIdx.y * 8 + wave;     // 0..15
  int x0 = xt * 16;
  int m0 = 8 * g;

  // A[m][k] = Lt[y][x0+m][k], K = C = 32 exactly; loop-invariant
  v16h a;
  const _Float16* arow = Lt + ((size_t)y * WW + (x0 + mn)) * CC;
#pragma unroll
  for (int r = 0; r < 8; ++r) {
    int k = klocA(r, g);
    a[2 * r] = arow[k];
    a[2 * r + 1] = arow[k + 1];
  }

  // per-lane base for the scatter: element vr -> (x = x0+m0+vr, d = d0-vr);
  // consecutive vr advance the address by exactly (DPAD-1) elements.
  _Float16* pbase =
      vol + ((size_t)(y + 1) * WP + (x0 + m0 + 1)) * DPAD + (mn - x0 - m0 + 200);

  for (int nt = 0; nt < 14; ++nt) {   // xr tiles covering [x0-208, x0+16)
    int xr0 = x0 - 208 + nt * 16;
    if (xr0 + 15 < 0) continue;       // wave-uniform skip (all-zero band)
    int xr = xr0 + mn;
    int xrc = xr < 0 ? 0 : xr;        // keep pointer arithmetic in-bounds
    const _Float16* brow = Rt + ((size_t)y * WW + xrc) * CC;
    v16h b;
#pragma unroll
    for (int r = 0; r < 8; ++r) {
      int k = klocB(r, g);
      b[2 * r]     = (xr >= 0) ? brow[k]     : (_Float16)0;
      b[2 * r + 1] = (xr >= 0) ? brow[k + 1] : (_Float16)0;
    }
    v8f c = {};
    c = __builtin_amdgcn_wmma_f32_16x16x32_f16(false, a, false, b, (short)0, c,
                                               false, false);
    int d0 = xr0 + mn - x0 - m0 + 200;
    _Float16* p = pbase + xr0;
#pragma unroll
    for (int vr = 0; vr < 8; ++vr) {
      int d = d0 - vr;
      if (d >= 0 && d < DD) p[vr * (DPAD - 1)] = (_Float16)c[vr];
    }
  }
}

// ------------------ implicit-GEMM 3x3 conv, f16 in / f32 acc, fused BN+ReLU
// one wave: MT x NT grid of 16x16 tiles (MT out-ch tiles, NT x-pixel tiles)
template <int MT, int NT>
__global__ void conv_wmma_k(const _Float16* __restrict__ ain,
                            const _Float16* __restrict__ W2,
                            const float* __restrict__ sc,
                            const float* __restrict__ sh,
                            _Float16* __restrict__ aout,
                            float* __restrict__ fout, int Cpad, int CoutPad,
                            int relu) {
  constexpr int NX = WW / (16 * NT);  // x-tile groups per row (power of 2)
  int wave = threadIdx.x >> 5, lane = threadIdx.x & 31;
  int g = lane >> 4, mn = lane & 15;
  int task = blockIdx.x * 8 + wave;
  int xt = task % NX;                 // 16*NT-pixel tile
  int y = (task / NX) % HH;
  int otg = task / (NX * HH);         // group of MT out-channel tiles
  int x0 = xt * (16 * NT);
  int S = Cpad >> 5;                  // 32-chunks per (ky,kx) tap
  int krow = 9 * Cpad;

  const _Float16* wrow[MT];
#pragma unroll
  for (int mt = 0; mt < MT; ++mt)
    wrow[mt] = W2 + (size_t)((otg * MT + mt) * 16 + mn) * krow;  // lane = M row

  v8f acc[MT][NT];
  v8f zz = {};
#pragma unroll
  for (int mt = 0; mt < MT; ++mt)
#pragma unroll
    for (int nt = 0; nt < NT; ++nt) acc[mt][nt] = zz;

  for (int ky = 0; ky < 3; ++ky) {
    for (int kx = 0; kx < 3; ++kx) {
      // padded coords: interior pixel (y,x) lives at (y+1, x+1)
      const _Float16* brow0 =
          ain + ((size_t)(y + ky) * WP + (x0 + mn + kx)) * Cpad;
      const _Float16* wtap[MT];
#pragma unroll
      for (int mt = 0; mt < MT; ++mt) wtap[mt] = wrow[mt] + (ky * 3 + kx) * Cpad;
      for (int cb = 0; cb < S; ++cb) {
        int kb0 = cb * 32;
        v16h a[MT], b[NT];
#pragma unroll
        for (int r = 0; r < 8; ++r) {
          int ka = klocA(r, g) + kb0;
#pragma unroll
          for (int mt = 0; mt < MT; ++mt) {
            a[mt][2 * r] = wtap[mt][ka];
            a[mt][2 * r + 1] = wtap[mt][ka + 1];
          }
          int kb = klocB(r, g) + kb0;
#pragma unroll
          for (int nt = 0; nt < NT; ++nt) {
            const _Float16* bp = brow0 + (size_t)(16 * nt) * Cpad;
            b[nt][2 * r] = bp[kb];
            b[nt][2 * r + 1] = bp[kb + 1];
          }
        }
#pragma unroll
        for (int mt = 0; mt < MT; ++mt)
#pragma unroll
          for (int nt = 0; nt < NT; ++nt)
            acc[mt][nt] = __builtin_amdgcn_wmma_f32_16x16x32_f16(
                false, a[mt], false, b[nt], (short)0, acc[mt][nt], false,
                false);
      }
    }
  }

  // epilogue: BN fold + ReLU; a lane's 8 C-elements are contiguous channels
  // -> pack to one 16B f16 store (interior) or f32 scalar (final layer, o==0)
#pragma unroll
  for (int mt = 0; mt < MT; ++mt) {
    int obase = (otg * MT + mt) * 16 + 8 * g;  // multiple of 8 -> 16B aligned
    float4 s0 = *(const float4*)(sc + obase);
    float4 s1 = *(const float4*)(sc + obase + 4);
    float4 h0 = *(const float4*)(sh + obase);
    float4 h1 = *(const float4*)(sh + obase + 4);
    float ss[8] = {s0.x, s0.y, s0.z, s0.w, s1.x, s1.y, s1.z, s1.w};
    float tt[8] = {h0.x, h0.y, h0.z, h0.w, h1.x, h1.y, h1.z, h1.w};
#pragma unroll
    for (int nt = 0; nt < NT; ++nt) {
      int x = x0 + nt * 16 + mn;
      if (fout) {
        if (obase == 0) fout[y * WW + x] = ss[0] * acc[mt][nt][0] + tt[0];
      } else {
        v8h hv;
#pragma unroll
        for (int vr = 0; vr < 8; ++vr) {
          float val = ss[vr] * acc[mt][nt][vr] + tt[vr];
          if (relu) val = fmaxf(val, 0.f);
          hv[vr] = (_Float16)val;
        }
        *(v8h*)(aout + ((size_t)(y + 1) * WP + (x + 1)) * CoutPad + obase) = hv;
      }
    }
  }
}

// ---------------------------------------------------------------- launch
extern "C" void kernel_launch(void* const* d_in, const int* in_sizes, int n_in,
                              void* d_out, int out_size, void* d_ws,
                              size_t ws_size, hipStream_t stream) {
  const float* left = (const float*)d_in[0];
  const float* right = (const float*)d_in[1];
  const float *w1 = (const float*)d_in[2], *b1 = (const float*)d_in[3],
              *g1 = (const float*)d_in[4], *be1 = (const float*)d_in[5],
              *m1 = (const float*)d_in[6], *v1 = (const float*)d_in[7];
  const float *w2 = (const float*)d_in[8], *b2 = (const float*)d_in[9],
              *g2 = (const float*)d_in[10], *be2 = (const float*)d_in[11],
              *m2 = (const float*)d_in[12], *v2 = (const float*)d_in[13];
  const float *w3 = (const float*)d_in[14], *b3 = (const float*)d_in[15],
              *g3 = (const float*)d_in[16], *be3 = (const float*)d_in[17],
              *m3 = (const float*)d_in[18], *v3 = (const float*)d_in[19];
  const float *w4 = (const float*)d_in[20], *b4 = (const float*)d_in[21];
  float* out = (float*)d_out;

  // workspace layout (256B aligned slabs); zeroed region first
  char* base = (char*)d_ws;
  size_t off = 0;
  auto alloc = [&](size_t bytes) {
    size_t o = off;
    off = (off + bytes + 255) & ~(size_t)255;
    return o;
  };
  const size_t PIX = (size_t)HP * WP;  // 33540 padded pixels
  size_t o_vol = alloc(PIX * DPAD * 2);   // f16 [130][258][224]
  size_t o_a1 = alloc(PIX * 64 * 2);      // f16 [130][258][64]
  size_t o_a2 = alloc(PIX * 32 * 2);
  size_t o_a3 = alloc(PIX * 32 * 2);      // 8 real ch + zero pad
  size_t zero_bytes = off;                // everything above must start zeroed
  size_t o_Lt = alloc((size_t)HH * WW * CC * 2);
  size_t o_Rt = alloc((size_t)HH * WW * CC * 2);
  size_t o_W1 = alloc((size_t)64 * 9 * DPAD * 2);
  size_t o_W2 = alloc((size_t)32 * 9 * 64 * 2);
  size_t o_W3 = alloc((size_t)16 * 9 * 32 * 2);
  size_t o_W4 = alloc((size_t)16 * 9 * 32 * 2);
  size_t o_s1 = alloc(64 * 4), o_t1 = alloc(64 * 4);
  size_t o_s2 = alloc(32 * 4), o_t2 = alloc(32 * 4);
  size_t o_s3 = alloc(16 * 4), o_t3 = alloc(16 * 4);
  size_t o_s4 = alloc(16 * 4), o_t4 = alloc(16 * 4);
  (void)ws_size;

  _Float16* vol = (_Float16*)(base + o_vol);
  _Float16* a1 = (_Float16*)(base + o_a1);
  _Float16* a2 = (_Float16*)(base + o_a2);
  _Float16* a3 = (_Float16*)(base + o_a3);
  _Float16* Lt = (_Float16*)(base + o_Lt);
  _Float16* Rt = (_Float16*)(base + o_Rt);
  _Float16* W1p = (_Float16*)(base + o_W1);
  _Float16* W2p = (_Float16*)(base + o_W2);
  _Float16* W3p = (_Float16*)(base + o_W3);
  _Float16* W4p = (_Float16*)(base + o_W4);
  float *s1 = (float*)(base + o_s1), *t1 = (float*)(base + o_t1);
  float *s2 = (float*)(base + o_s2), *t2 = (float*)(base + o_t2);
  float *s3 = (float*)(base + o_s3), *t3 = (float*)(base + o_t3);
  float *s4 = (float*)(base + o_s4), *t4 = (float*)(base + o_t4);

  const int TB = 256;

  // 1) zero padded activation slabs (borders/padding must be exactly 0)
  zero_kernel<<<2048, TB, 0, stream>>>((float4*)base, (long)(zero_bytes / 16));

  // 2) transpose + f16-convert left/right
  {
    int total = CC * HH * WW;
    cvt_lr<<<(total + TB - 1) / TB, TB, 0, stream>>>(left, right, Lt, Rt);
  }

  // 3) weight repack + BN folding
  prep_w<<<512, TB, 0, stream>>>(w1, b1, g1, be1, m1, v1, W1p, s1, t1, 64, 64,
                                 DD, DPAD, 1);
  prep_w<<<128, TB, 0, stream>>>(w2, b2, g2, be2, m2, v2, W2p, s2, t2, 32, 32,
                                 64, 64, 1);
  prep_w<<<32, TB, 0, stream>>>(w3, b3, g3, be3, m3, v3, W3p, s3, t3, 8, 16,
                                32, 32, 1);
  prep_w<<<32, TB, 0, stream>>>(w4, b4, nullptr, nullptr, nullptr, nullptr,
                                W4p, s4, t4, 1, 16, 8, 32, 0);

  // 4) cost volume via banded WMMA GEMM: grid (y=128, xt-groups=2) x 8 waves
  corr_wmma<<<dim3(128, 2), TB, 0, stream>>>(Lt, Rt, vol);

  // 5) conv stack: tasks = (Opad/(16*MT)) * 128 rows * NX x-tiles, 8 waves/blk
  conv_wmma_k<2, 4><<<(2 * 128 * 4) / 8, TB, 0, stream>>>(vol, W1p, s1, t1, a1,
                                                          nullptr, DPAD, 64, 1);
  conv_wmma_k<2, 4><<<(1 * 128 * 4) / 8, TB, 0, stream>>>(a1, W2p, s2, t2, a2,
                                                          nullptr, 64, 32, 1);
  conv_wmma_k<1, 2><<<(1 * 128 * 8) / 8, TB, 0, stream>>>(a2, W3p, s3, t3, a3,
                                                          nullptr, 32, 32, 1);
  conv_wmma_k<1, 2><<<(1 * 128 * 8) / 8, TB, 0, stream>>>(a3, W4p, s4, t4,
                                                          nullptr, out, 32, 0,
                                                          0);
  (void)in_sizes;
  (void)n_in;
  (void)out_size;
}
